// EmbeddingMI1_17918603559314
// MI455X (gfx1250) — compile-verified, matrix operand order checked by reading, not run on hardware
//
#include <hip/hip_runtime.h>
#include <hip/hip_bf16.h>
#include <math.h>

// ---------------------------------------------------------------------------
// Problem constants (match reference: B, IN, D, H, S)
// ---------------------------------------------------------------------------
#define BATCH   128
#define IN_DIM  1024
#define DDIM    128
#define HDIM    512
#define SSAMP   256
#define MTOT    ((SSAMP + 1) * BATCH)   // 32896 rows: w_tilde flat + w
#define MTILES  (MTOT / 16)             // 2056 (divides exactly)

typedef __attribute__((ext_vector_type(16))) _Float16 v16h;
typedef __attribute__((ext_vector_type(8)))  float    v8f;

__device__ __forceinline__ float softplusf(float t) {
  // numerically stable: max(t,0) + log1p(exp(-|t|))
  return fmaxf(t, 0.0f) + log1pf(expf(-fabsf(t)));
}

// ---------------------------------------------------------------------------
// CDNA5 async global->LDS copy (CDNA5 ISA). vdst = LDS byte address,
// vaddr = 64-bit global address. Tracked by ASYNCcnt.
// ---------------------------------------------------------------------------
__device__ __forceinline__ void async_load_b128(unsigned lds_byte, const void* g) {
  asm volatile("global_load_async_to_lds_b128 %0, %1, off"
               :: "v"(lds_byte), "v"((unsigned long long)(uintptr_t)g)
               : "memory");
}
__device__ __forceinline__ void wait_async_le1() {
  asm volatile("s_wait_asynccnt 1" ::: "memory");
}
__device__ __forceinline__ void wait_async_le0() {
  asm volatile("s_wait_asynccnt 0" ::: "memory");
}

// Per-lane share of a 16x32 fp16 fragment gathered from a row-major fp32 row.
// Per CDNA5 ISA 7.12.2 (16-bit A 16x32): lane<16 holds K {0..7, 16..23},
// lane>=16 holds K {8..15, 24..31}.
__device__ __forceinline__ v16h gather_frag_f32(const float* __restrict__ rowptr,
                                                int k0, int koff) {
  const float* p0 = rowptr + k0 + koff;
  const float* p1 = p0 + 16;
  v16h f;
#pragma unroll
  for (int i = 0; i < 8; ++i) f[i] = (_Float16)p0[i];
#pragma unroll
  for (int i = 0; i < 8; ++i) f[8 + i] = (_Float16)p1[i];
  return f;
}

// ---------------------------------------------------------------------------
// Kernel 0: convert W1[:, D:2D] to fp16, row-major [H][D] (contiguous per h)
// ---------------------------------------------------------------------------
__global__ void convert_w1_kernel(const float* __restrict__ W1,
                                  _Float16* __restrict__ W1h) {
  const int idx = blockIdx.x * blockDim.x + threadIdx.x;   // HDIM*DDIM
  const int h = idx >> 7;
  const int k = idx & (DDIM - 1);
  W1h[idx] = (_Float16)W1[(size_t)h * (2 * DDIM) + DDIM + k];
}

// ---------------------------------------------------------------------------
// Kernel 1: zpart[b,h] = b1[h] + sum_d z[b,d] * W1[h,d]   (fp32, tiny GEMV)
// ---------------------------------------------------------------------------
__global__ void zpart_kernel(const float* __restrict__ z,
                             const float* __restrict__ W1,
                             const float* __restrict__ b1,
                             float* __restrict__ zpart) {
  const int idx = blockIdx.x * blockDim.x + threadIdx.x;   // BATCH*HDIM threads
  const int b = idx >> 9;            // /HDIM
  const int h = idx & (HDIM - 1);
  const float4* zr = (const float4*)(z + (size_t)b * DDIM);
  const float4* wr = (const float4*)(W1 + (size_t)h * (2 * DDIM)); // first D cols
  float acc = b1[h];
#pragma unroll 4
  for (int q = 0; q < DDIM / 4; ++q) {
    const float4 a = zr[q];
    const float4 c = wr[q];
    acc += a.x * c.x + a.y * c.y + a.z * c.z + a.w * c.w;
  }
  zpart[idx] = acc;
}

// ---------------------------------------------------------------------------
// Kernel 2: lp[b] = log p(x|w) - log p(w|x)   (one 256-thread block per b)
// ---------------------------------------------------------------------------
__global__ void bern_kernel(const float* __restrict__ x,
                            const float* __restrict__ w,
                            const float* __restrict__ W,
                            const float* __restrict__ bvec,
                            const float* __restrict__ cvec,
                            float* __restrict__ lp) {
  __shared__ float xs[IN_DIM];    // 4KB: x row, reused heavily in phase 2
  __shared__ float wv[DDIM];
  __shared__ float red[8];
  const int b = blockIdx.x;
  const int tid = threadIdx.x;
  const float* xr = x + (size_t)b * IN_DIM;
  for (int i = tid; i < IN_DIM; i += 256) xs[i] = xr[i];
  if (tid < DDIM) wv[tid] = w[(size_t)b * DDIM + tid];
  __syncthreads();

  float acc = 0.0f;

  // + log p(x|w): logits_i = c[i] + w[b,:]·W[i,:]   (W is [IN, D] row-major)
  for (int i = tid; i < IN_DIM; i += 256) {
    const float4* Wr = (const float4*)(W + (size_t)i * DDIM);
    float logit = cvec[i];
#pragma unroll 4
    for (int q = 0; q < DDIM / 4; ++q) {
      const float4 ww = Wr[q];
      logit += wv[4 * q] * ww.x + wv[4 * q + 1] * ww.y +
               wv[4 * q + 2] * ww.z + wv[4 * q + 3] * ww.w;
    }
    acc += xs[i] * logit - softplusf(logit);
  }

  // - log p(w|x): logits_d = b[d] + x[b,:]·W[:,d]
  if (tid < DDIM) {
    const int d = tid;
    float logit = bvec[d];
    for (int i = 0; i < IN_DIM; ++i)
      logit += xs[i] * W[(size_t)i * DDIM + d];    // coalesced across d
    acc -= wv[d] * logit - softplusf(logit);
  }

  // block reduction (wave32 shuffles, then LDS across 8 waves)
#pragma unroll
  for (int off = 16; off > 0; off >>= 1) acc += __shfl_xor(acc, off, 32);
  if ((tid & 31) == 0) red[tid >> 5] = acc;
  __syncthreads();
  if (tid == 0) {
    float t = 0.0f;
#pragma unroll
    for (int i = 0; i < 8; ++i) t += red[i];
    lp[b] = t;
  }
}

// ---------------------------------------------------------------------------
// Kernel 3: energy GEMM via v_wmma_f32_16x16x32_f16.
//   rows m = s*B + b: m < S*B -> w_tilde flat; else -> w (f_zw row block)
//   f[m] = b2 + sum_h relu(zpart[b,h] + A[m,:]·W1[h,D:]) * W2[h]
// One wave per 16-row M-tile (8 waves/block). The fp16 B panel (16 h-rows x
// 128 k) is double-buffered in LDS via async global->LDS copies shared by all
// 8 waves of the block.
// ---------------------------------------------------------------------------
__global__ void __launch_bounds__(256)
energy_kernel(const float* __restrict__ w_tilde,
              const float* __restrict__ w,
              const _Float16* __restrict__ W1h,   // [H][D] fp16
              const float* __restrict__ W2,
              const float* __restrict__ zpart,
              const float* __restrict__ b2,
              float* __restrict__ f) {
  __shared__ __align__(16) _Float16 panel[2][16 * DDIM];   // 2 x 4KB

  const int tid   = threadIdx.x;
  const int lane  = tid & 31;
  const int wid   = tid >> 5;
  const int mtile = blockIdx.x * 8 + wid;      // 0..2055
  const int m0    = mtile * 16;
  const int r     = lane & 15;                 // A row-in-tile / B column (h)
  const int hi    = lane >> 4;                 // half-wave selector
  const int koff  = hi * 8;                    // K sub-offset per ISA layout
  const int m     = m0 + r;

  // Tiles never straddle the w_tilde/w boundary (32768 % 16 == 0).
  const float* Arow = (m0 < SSAMP * BATCH)
      ? (w_tilde + (size_t)m * DDIM)
      : (w + (size_t)(m - SSAMP * BATCH) * DDIM);

  v16h afrag[4];
#pragma unroll
  for (int kt = 0; kt < 4; ++kt) afrag[kt] = gather_frag_f32(Arow, kt * 32, koff);

  // Async staging: thread tid copies bytes [tid*16, tid*16+16) of the 4KB panel.
  // LDS flat address truncates to the LDS byte offset (ISA 10.2).
  const unsigned lds0 = (unsigned)(uintptr_t)(&panel[0][0]) + (unsigned)tid * 16u;
  const unsigned lds1 = (unsigned)(uintptr_t)(&panel[1][0]) + (unsigned)tid * 16u;
  const char* gpan = (const char*)W1h;         // panel ht = gpan + ht*4096

  async_load_b128(lds0, gpan + (size_t)tid * 16);   // prologue: panel 0

  const int b_base = m0 & (BATCH - 1);         // batch index of row 0 in tile

  float facc[8];
#pragma unroll
  for (int v = 0; v < 8; ++v) facc[v] = 0.0f;

  for (int ht = 0; ht < HDIM / 16; ++ht) {
    if (ht + 1 < HDIM / 16) {
      // prefetch next panel into the other buffer, then wait for current one
      async_load_b128((ht & 1) ? lds0 : lds1,
                      gpan + (size_t)(ht + 1) * 4096 + (size_t)tid * 16);
      wait_async_le1();
    } else {
      wait_async_le0();
    }
    __syncthreads();                            // panel ht visible to all waves

    const _Float16* pan = &panel[ht & 1][0];
    v8f acc = {};
#pragma unroll
    for (int kt = 0; kt < 4; ++kt) {
      const _Float16* p0 = pan + r * DDIM + kt * 32 + koff;  // 16B aligned
      v16h bfrag;
#pragma unroll
      for (int i = 0; i < 8; ++i) bfrag[i] = p0[i];
#pragma unroll
      for (int i = 0; i < 8; ++i) bfrag[8 + i] = p0[16 + i];
      acc = __builtin_amdgcn_wmma_f32_16x16x32_f16(
          /*neg_a=*/false, afrag[kt], /*neg_b=*/false, bfrag,
          /*c_mod=*/(short)0, acc, /*reuse_a=*/false, /*reuse_b=*/false);
    }

    // Epilogue: C/D layout -> VGPR v holds row M = v + 8*hi, col N = lane&15.
    const int h = ht * 16 + r;
    const float w2v = W2[h];
#pragma unroll
    for (int v = 0; v < 8; ++v) {
      const float zp = zpart[(size_t)(b_base + v + 8 * hi) * HDIM + h];
      const float pre = acc[v] + zp;
      facc[v] += fmaxf(pre, 0.0f) * w2v;       // relu * W2
    }
    __syncthreads();                            // safe to overwrite buf[ht&1]
  }

  // Reduce over the 16 h-columns held across lanes of each half-wave.
#pragma unroll
  for (int v = 0; v < 8; ++v) {
    float t = facc[v];
    t += __shfl_xor(t, 1, 32);
    t += __shfl_xor(t, 2, 32);
    t += __shfl_xor(t, 4, 32);
    t += __shfl_xor(t, 8, 32);
    facc[v] = t;
  }
  if (r == 0) {
    const float bb2 = b2[0];
#pragma unroll
    for (int v = 0; v < 8; ++v) f[m0 + v + 8 * hi] = facc[v] + bb2;
  }
}

// ---------------------------------------------------------------------------
// Kernel 4: per-batch logsumexp over S, clamp, combine with lp.
// ---------------------------------------------------------------------------
__global__ void finalize_kernel(const float* __restrict__ f,
                                const float* __restrict__ lp,
                                float* __restrict__ out) {
  const int b = threadIdx.x;                   // 0..127
  float mx = -INFINITY;
  for (int s = 0; s < SSAMP; ++s) mx = fmaxf(mx, f[s * BATCH + b]);
  float sum = 0.0f;
  for (int s = 0; s < SSAMP; ++s) sum += expf(f[s * BATCH + b] - mx);
  const float logZ = mx + logf(sum) - logf((float)SSAMP)
                   + (float)DDIM * logf(2.0f);
  const float r = fminf(f[SSAMP * BATCH + b] - logZ, 0.0f);
  out[b] = -(lp[b] + r);
}

// ---------------------------------------------------------------------------
// Launch
// ---------------------------------------------------------------------------
extern "C" void kernel_launch(void* const* d_in, const int* in_sizes, int n_in,
                              void* d_out, int out_size, void* d_ws, size_t ws_size,
                              hipStream_t stream) {
  const float* x   = (const float*)d_in[0];
  // d_in[1] = y (unused by the loss; only used to produce z in setup)
  const float* w   = (const float*)d_in[2];
  const float* z   = (const float*)d_in[3];
  const float* wt  = (const float*)d_in[4];
  const float* W   = (const float*)d_in[5];
  const float* bv  = (const float*)d_in[6];
  const float* cv  = (const float*)d_in[7];
  const float* W1  = (const float*)d_in[8];
  const float* b1  = (const float*)d_in[9];
  const float* W2  = (const float*)d_in[10];
  const float* b2  = (const float*)d_in[11];
  float* out = (float*)d_out;

  char* ws = (char*)d_ws;
  size_t off = 0;
  float* zpart = (float*)(ws + off);  off += sizeof(float) * BATCH * HDIM;   // 256KB
  float* fbuf  = (float*)(ws + off);  off += sizeof(float) * MTOT;           // 128.5KB
  float* lp    = (float*)(ws + off);  off += sizeof(float) * BATCH;
  off = (off + 255) & ~(size_t)255;
  _Float16* W1h = (_Float16*)(ws + off);                                     // 128KB

  convert_w1_kernel<<<(HDIM * DDIM) / 256, 256, 0, stream>>>(W1, W1h);
  zpart_kernel<<<(BATCH * HDIM) / 256, 256, 0, stream>>>(z, W1, b1, zpart);
  bern_kernel<<<BATCH, 256, 0, stream>>>(x, w, W, bv, cv, lp);
  energy_kernel<<<MTILES / 8, 256, 0, stream>>>(wt, w, W1h, W2, zpart, b2, fbuf);
  finalize_kernel<<<1, BATCH, 0, stream>>>(fbuf, lp, out);
}
